// MoE_Block_39444979646723
// MI455X (gfx1250) — matrix operand order repaired and checked by reading, not compiled, not run on hardware
//
#include <hip/hip_runtime.h>
#include <hip/hip_bf16.h>

// Problem constants (match reference: B,S,D,E,DOUT,K)
#define BB    4
#define SS    2048
#define DD    1024
#define EE    8
#define DOUT  1024
#define KK    2
#define TT    (BB * SS)        // 8192 tokens
#define CAP   TT               // per-expert capacity

typedef __attribute__((ext_vector_type(16))) __bf16        v16bf;
typedef __attribute__((ext_vector_type(2)))  __bf16        v2bf;
typedef __attribute__((ext_vector_type(8)))  float         v8f;
typedef __attribute__((ext_vector_type(8)))  unsigned int  v8u;
typedef int v4i __attribute__((__vector_size__(16)));

#define AS1 __attribute__((address_space(1)))
#define AS3 __attribute__((address_space(3)))

#if __has_builtin(__builtin_amdgcn_global_load_async_to_lds_b128)
#define ASYNC_STAGE 1
__device__ __forceinline__ void async_copy16(const void* g, void* l) {
  __builtin_amdgcn_global_load_async_to_lds_b128((AS1 v4i*)g, (AS3 v4i*)l, 0, 0);
}
#endif

__device__ __forceinline__ unsigned int pack2bf(float lo, float hi) {
#if __has_builtin(__builtin_amdgcn_cvt_pk_bf16_f32)
  v2bf v = __builtin_amdgcn_cvt_pk_bf16_f32(lo, hi);
  return __builtin_bit_cast(unsigned int, v);
#else
  unsigned int ul = __float_as_uint(lo), uh = __float_as_uint(hi);
  ul += 0x7FFFu + ((ul >> 16) & 1u);
  uh += 0x7FFFu + ((uh >> 16) & 1u);
  return (ul >> 16) | (uh & 0xFFFF0000u);
#endif
}

// ---------------------------------------------------------------------------
// Kernel 0: zero output (atomic-accumulated across graph replays) + ws header
// ---------------------------------------------------------------------------
__global__ __launch_bounds__(256) void moe_zero(float* __restrict__ y,
                                                int* __restrict__ hdr) {
  size_t i = (size_t)blockIdx.x * 2048 + (size_t)threadIdx.x * 8;
  float4 z = make_float4(0.f, 0.f, 0.f, 0.f);
  float4* p = (float4*)(y + i);
  p[0] = z; p[1] = z;
  if (blockIdx.x == 0 && threadIdx.x < 64) hdr[threadIdx.x] = 0;
}

// ---------------------------------------------------------------------------
// Kernel 0b: f32 -> bf16 pre-conversion of x and We into workspace
// ---------------------------------------------------------------------------
__global__ __launch_bounds__(256) void moe_convert(
    const float* __restrict__ x, const float* __restrict__ We,
    unsigned short* __restrict__ xbf, unsigned short* __restrict__ Webf) {
  size_t idx = ((size_t)blockIdx.x * 256 + threadIdx.x) * 8;
  const float* src; unsigned short* dst; size_t off;
  if (idx < (size_t)TT * DD) { src = x;  dst = xbf;  off = idx; }
  else                       { src = We; dst = Webf; off = idx - (size_t)TT * DD; }
  float4 a = *(const float4*)(src + off);
  float4 b = *(const float4*)(src + off + 4);
  uint4 o;
  o.x = pack2bf(a.x, a.y); o.y = pack2bf(a.z, a.w);
  o.z = pack2bf(b.x, b.y); o.w = pack2bf(b.z, b.w);
  *(uint4*)(dst + off) = o;
}

// ---------------------------------------------------------------------------
// Kernel 1: router. One wave32 per token, 8 tokens per block.
// ---------------------------------------------------------------------------
__global__ __launch_bounds__(256) void moe_router(
    const float* __restrict__ x, const float* __restrict__ Wr,
    const float* __restrict__ br,
    int* __restrict__ counts, float* __restrict__ sum_probs,
    int* __restrict__ etok, float* __restrict__ egate) {
  __shared__ float sblk[EE];
  const int tid  = threadIdx.x;
  const int wave = tid >> 5, lane = tid & 31;
  if (tid < EE) sblk[tid] = 0.f;
  __syncthreads();

  const int t = blockIdx.x * 8 + wave;
  float acc[EE];
#pragma unroll
  for (int e = 0; e < EE; ++e) acc[e] = 0.f;

  const float* xr = x + (size_t)t * DD;
  for (int d = lane; d < DD; d += 32) {
    float xv = xr[d];
    const float4* w4 = (const float4*)(Wr + (size_t)d * EE);
    float4 w0 = w4[0], w1 = w4[1];
    acc[0] += xv * w0.x; acc[1] += xv * w0.y;
    acc[2] += xv * w0.z; acc[3] += xv * w0.w;
    acc[4] += xv * w1.x; acc[5] += xv * w1.y;
    acc[6] += xv * w1.z; acc[7] += xv * w1.w;
  }
#pragma unroll
  for (int off = 16; off > 0; off >>= 1) {
#pragma unroll
    for (int e = 0; e < EE; ++e) acc[e] += __shfl_xor(acc[e], off, 32);
  }

  if (lane == 0) {
    float p[EE];
    float mx = -1e30f;
#pragma unroll
    for (int e = 0; e < EE; ++e) { p[e] = acc[e] + br[e]; mx = fmaxf(mx, p[e]); }
    float s = 0.f;
#pragma unroll
    for (int e = 0; e < EE; ++e) { p[e] = __expf(p[e] - mx); s += p[e]; }
    float inv = 1.f / s;
#pragma unroll
    for (int e = 0; e < EE; ++e) p[e] *= inv;

    int i0 = 0; float g0 = p[0];
#pragma unroll
    for (int e = 1; e < EE; ++e) if (p[e] > g0) { g0 = p[e]; i0 = e; }
    int i1 = (i0 == 0) ? 1 : 0; float g1 = p[i1];
#pragma unroll
    for (int e = 0; e < EE; ++e) if (e != i0 && p[e] > g1) { g1 = p[e]; i1 = e; }
    float nrm = 1.f / (g0 + g1 + 1e-9f);
    g0 *= nrm; g1 *= nrm;

#pragma unroll
    for (int e = 0; e < EE; ++e) atomicAdd(&sblk[e], p[e]);

    int p0 = atomicAdd(&counts[i0], 1);
    etok[i0 * CAP + p0] = t; egate[i0 * CAP + p0] = g0;
    int p1 = atomicAdd(&counts[i1], 1);
    etok[i1 * CAP + p1] = t; egate[i1 * CAP + p1] = g1;
  }
  __syncthreads();
  if (tid < EE) atomicAdd(&sum_probs[tid], sblk[tid]);
}

// ---------------------------------------------------------------------------
// Shared WMMA compute step: one 32-wide K slice on a 128x128 block tile.
// Wave (wm,wn) owns a 32(M)x64(N) slab: 2 A-frags x 4 B-frags, B reused.
// ---------------------------------------------------------------------------
__device__ __forceinline__ void frag_step(const unsigned short* __restrict__ Xb,
                                          const unsigned short* __restrict__ Wb,
                                          int wm, int wn, int lan, int half,
                                          v8f accv[2][4]) {
  constexpr int ST = 40;
  v16bf afrag[2];
#pragma unroll
  for (int mf = 0; mf < 2; ++mf) {
    const char* row = (const char*)(Xb + (wm * 32 + mf * 16 + lan) * ST);
    uint4 lo = *(const uint4*)(row + half * 16);
    uint4 hi = *(const uint4*)(row + 32 + half * 16);
    v8u u;
    u[0] = lo.x; u[1] = lo.y; u[2] = lo.z; u[3] = lo.w;
    u[4] = hi.x; u[5] = hi.y; u[6] = hi.z; u[7] = hi.w;
    afrag[mf] = __builtin_bit_cast(v16bf, u);
  }
#pragma unroll
  for (int f = 0; f < 4; ++f) {
    const char* row = (const char*)(Wb + (wn * 64 + f * 16 + lan) * ST);
    uint4 lo = *(const uint4*)(row + half * 16);
    uint4 hi = *(const uint4*)(row + 32 + half * 16);
    v8u u;
    u[0] = lo.x; u[1] = lo.y; u[2] = lo.z; u[3] = lo.w;
    u[4] = hi.x; u[5] = hi.y; u[6] = hi.z; u[7] = hi.w;
    v16bf bfrag = __builtin_bit_cast(v16bf, u);
    accv[0][f] = __builtin_amdgcn_wmma_f32_16x16x32_bf16(
        false, afrag[0], false, bfrag, (short)0, accv[0][f], false, false);
    accv[1][f] = __builtin_amdgcn_wmma_f32_16x16x32_bf16(
        false, afrag[1], false, bfrag, (short)0, accv[1][f], false, false);
  }
}

__device__ __forceinline__ void epilogue_scatter(
    const v8f accv[2][4], const float* __restrict__ be,
    const int* __restrict__ tokS, const float* __restrict__ gateS,
    float* __restrict__ y, int e, int n0, int mBase, int cnt,
    int wm, int wn, int lan, int half) {
  float bias[4];
#pragma unroll
  for (int f = 0; f < 4; ++f)
    bias[f] = be[(size_t)e * DOUT + n0 + wn * 64 + f * 16 + lan];
#pragma unroll
  for (int mf = 0; mf < 2; ++mf) {
#pragma unroll
    for (int r = 0; r < 8; ++r) {
      const int mloc = wm * 32 + mf * 16 + half * 8 + r;  // C/D: vgpr r -> M = r + 8*(lane/16)
      const bool valid = (mBase + mloc) < cnt;
      const float g = gateS[mloc];
      float* yrow = y + (size_t)tokS[mloc] * DOUT + n0 + wn * 64 + lan;
#pragma unroll
      for (int f = 0; f < 4; ++f) {
        if (valid) {
          float val = (accv[mf][f][r] + bias[f]) * g;
          __hip_atomic_fetch_add(yrow + f * 16, val,
                                 __ATOMIC_RELAXED, __HIP_MEMORY_SCOPE_AGENT);
        }
      }
    }
  }
}

// ---------------------------------------------------------------------------
// Kernel 2a (fast path): gathered GEMM on pre-converted bf16 data.
// X tile staged global->LDS with async-to-LDS when available; W tile
// transposed through VGPRs with v_perm packing.
// ---------------------------------------------------------------------------
__global__ __launch_bounds__(256) void moe_expert_gemm_bf16(
    const unsigned short* __restrict__ xbf, const unsigned short* __restrict__ Webf,
    const float* __restrict__ be,
    const int* __restrict__ counts, const int* __restrict__ etok,
    const float* __restrict__ egate, float* __restrict__ y) {
  constexpr int BM = 128, BN = 128, ST = 40;
  __shared__ unsigned short Xl[2][BM * ST];   // [m][k] bf16
  __shared__ unsigned short Wl[2][BN * ST];   // [n][k] bf16 (transposed)
  __shared__ int   tokS[BM];
  __shared__ float gateS[BM];

  const int e     = blockIdx.x >> 6;
  const int mt    = blockIdx.x & 63;
  const int cnt   = counts[e];
  const int mBase = mt * BM;
  if (mBase >= cnt) return;
  const int n0 = blockIdx.y * BN;

  const int tid  = threadIdx.x;
  const int wave = tid >> 5, lane = tid & 31;
  const int wm = wave & 3, wn = wave >> 2;
  const int half = lane >> 4, lan = lane & 15;

  if (tid < BM) {
    int gr = mBase + tid;
    bool v = gr < cnt;
    tokS[tid]  = v ? etok[e * CAP + gr]  : 0;
    gateS[tid] = v ? egate[e * CAP + gr] : 0.f;
  }
  __syncthreads();

  v8f accv[2][4];
#pragma unroll
  for (int mf = 0; mf < 2; ++mf)
#pragma unroll
    for (int f = 0; f < 4; ++f)
#pragma unroll
      for (int r = 0; r < 8; ++r) accv[mf][f][r] = 0.f;

  const unsigned short* WeE = Webf + (size_t)e * DD * DOUT;

  // X staging: row = tid/2, 16-elem segment = (tid&1)*16, two 16B chunks
  const int xrow = tid >> 1, xseg = (tid & 1) * 16;
  const unsigned short* xsrc = xbf + (size_t)tokS[xrow] * DD + xseg;
  unsigned short* xdstA = &Xl[0][xrow * ST + xseg];
  unsigned short* xdstB = &Xl[1][xrow * ST + xseg];

  auto stage_x = [&](int k0, int buf) {
    unsigned short* d = buf ? xdstB : xdstA;
#ifdef ASYNC_STAGE
    async_copy16(xsrc + k0, d);
    async_copy16(xsrc + k0 + 8, d + 8);
#else
    ((uint4*)d)[0] = *(const uint4*)(xsrc + k0);
    ((uint4*)d)[1] = *(const uint4*)(xsrc + k0 + 8);
#endif
  };

  // W staging: thread i (two sub-iters) -> k-pair 2*(i>>5), n4 = (i&31)*4
  uint2 wa[2], wb[2];
  auto load_w = [&](int k0) {
#pragma unroll
    for (int it = 0; it < 2; ++it) {
      int i  = tid + 256 * it;
      int k  = (i >> 5) << 1;
      int n4 = (i & 31) << 2;
      wa[it] = *(const uint2*)(WeE + (size_t)(k0 + k) * DOUT + n0 + n4);
      wb[it] = *(const uint2*)(WeE + (size_t)(k0 + k + 1) * DOUT + n0 + n4);
    }
  };
  auto store_w = [&](int buf) {
#pragma unroll
    for (int it = 0; it < 2; ++it) {
      int i  = tid + 256 * it;
      int k  = (i >> 5) << 1;
      int n4 = (i & 31) << 2;
      unsigned short* w = &Wl[buf][0];
      // pack (row k, row k+1) bf16 pairs with v_perm_b32
      *(unsigned int*)(w + (n4 + 0) * ST + k) = __builtin_amdgcn_perm(wb[it].x, wa[it].x, 0x05040100u);
      *(unsigned int*)(w + (n4 + 1) * ST + k) = __builtin_amdgcn_perm(wb[it].x, wa[it].x, 0x07060302u);
      *(unsigned int*)(w + (n4 + 2) * ST + k) = __builtin_amdgcn_perm(wb[it].y, wa[it].y, 0x05040100u);
      *(unsigned int*)(w + (n4 + 3) * ST + k) = __builtin_amdgcn_perm(wb[it].y, wa[it].y, 0x07060302u);
    }
  };

  constexpr int NT = DD / 32;
  stage_x(0, 0);
  load_w(0);
  store_w(0);

  for (int itk = 0; itk < NT; ++itk) {
    const int cur = itk & 1;
#ifdef ASYNC_STAGE
    asm volatile("s_wait_asynccnt 0" ::: "memory");
#endif
    __syncthreads();                               // buf[cur] published
    if (itk + 1 < NT) {
      stage_x((itk + 1) * 32, cur ^ 1);            // async: streams during compute
      load_w((itk + 1) * 32);
    }
    frag_step(&Xl[cur][0], &Wl[cur][0], wm, wn, lan, half, accv);
    if (itk + 1 < NT) store_w(cur ^ 1);
  }

  epilogue_scatter(accv, be, tokS, gateS, y, e, n0, mBase, cnt, wm, wn, lan, half);
}

// ---------------------------------------------------------------------------
// Kernel 2b (fallback if ws too small for bf16 copies): convert during staging
// ---------------------------------------------------------------------------
__global__ __launch_bounds__(256) void moe_expert_gemm_f32(
    const float* __restrict__ x, const float* __restrict__ We,
    const float* __restrict__ be,
    const int* __restrict__ counts, const int* __restrict__ etok,
    const float* __restrict__ egate, float* __restrict__ y) {
  constexpr int BM = 128, BN = 128, ST = 40;
  __shared__ unsigned short Xl[2][BM * ST];
  __shared__ unsigned short Wl[2][BN * ST];
  __shared__ int   tokS[BM];
  __shared__ float gateS[BM];

  const int e     = blockIdx.x >> 6;
  const int mt    = blockIdx.x & 63;
  const int cnt   = counts[e];
  const int mBase = mt * BM;
  if (mBase >= cnt) return;
  const int n0 = blockIdx.y * BN;

  const int tid  = threadIdx.x;
  const int wave = tid >> 5, lane = tid & 31;
  const int wm = wave & 3, wn = wave >> 2;
  const int half = lane >> 4, lan = lane & 15;

  if (tid < BM) {
    int gr = mBase + tid;
    bool v = gr < cnt;
    tokS[tid]  = v ? etok[e * CAP + gr]  : 0;
    gateS[tid] = v ? egate[e * CAP + gr] : 0.f;
  }
  __syncthreads();

  v8f accv[2][4];
#pragma unroll
  for (int mf = 0; mf < 2; ++mf)
#pragma unroll
    for (int f = 0; f < 4; ++f)
#pragma unroll
      for (int r = 0; r < 8; ++r) accv[mf][f][r] = 0.f;

  const float* WeE = We + (size_t)e * DD * DOUT;
  const int xrow = tid >> 1, xseg = (tid & 1) * 16;
  const float* xsrc = x + (size_t)tokS[xrow] * DD + xseg;

  float4 xg[4];
  float4 wg0[2], wg1[2];
  auto load_tile = [&](int k0) {
#pragma unroll
    for (int j = 0; j < 4; ++j) xg[j] = *(const float4*)(xsrc + k0 + 4 * j);
#pragma unroll
    for (int it = 0; it < 2; ++it) {
      int i  = tid + 256 * it;
      int k  = (i >> 5) << 1;
      int n4 = (i & 31) << 2;
      wg0[it] = *(const float4*)(WeE + (size_t)(k0 + k) * DOUT + n0 + n4);
      wg1[it] = *(const float4*)(WeE + (size_t)(k0 + k + 1) * DOUT + n0 + n4);
    }
  };
  auto store_tile = [&](int buf) {
    unsigned int* xd = (unsigned int*)(&Xl[buf][xrow * ST + xseg]);
    xd[0] = pack2bf(xg[0].x, xg[0].y); xd[1] = pack2bf(xg[0].z, xg[0].w);
    xd[2] = pack2bf(xg[1].x, xg[1].y); xd[3] = pack2bf(xg[1].z, xg[1].w);
    xd[4] = pack2bf(xg[2].x, xg[2].y); xd[5] = pack2bf(xg[2].z, xg[2].w);
    xd[6] = pack2bf(xg[3].x, xg[3].y); xd[7] = pack2bf(xg[3].z, xg[3].w);
#pragma unroll
    for (int it = 0; it < 2; ++it) {
      int i  = tid + 256 * it;
      int k  = (i >> 5) << 1;
      int n4 = (i & 31) << 2;
      unsigned short* w = &Wl[buf][0];
      *(unsigned int*)(w + (n4 + 0) * ST + k) = pack2bf(wg0[it].x, wg1[it].x);
      *(unsigned int*)(w + (n4 + 1) * ST + k) = pack2bf(wg0[it].y, wg1[it].y);
      *(unsigned int*)(w + (n4 + 2) * ST + k) = pack2bf(wg0[it].z, wg1[it].z);
      *(unsigned int*)(w + (n4 + 3) * ST + k) = pack2bf(wg0[it].w, wg1[it].w);
    }
  };

  constexpr int NT = DD / 32;
  load_tile(0);
  store_tile(0);
  for (int itk = 0; itk < NT; ++itk) {
    const int cur = itk & 1;
    __syncthreads();
    if (itk + 1 < NT) load_tile((itk + 1) * 32);
    frag_step(&Xl[cur][0], &Wl[cur][0], wm, wn, lan, half, accv);
    if (itk + 1 < NT) store_tile(cur ^ 1);
  }

  epilogue_scatter(accv, be, tokS, gateS, y, e, n0, mBase, cnt, wm, wn, lan, half);
}

// ---------------------------------------------------------------------------
// Kernel 3: aux loss scalar -> y[T*DOUT]
// ---------------------------------------------------------------------------
__global__ void moe_aux(const int* __restrict__ counts,
                        const float* __restrict__ sum_probs,
                        float* __restrict__ y) {
  if (threadIdx.x == 0 && blockIdx.x == 0) {
    float acc = 0.f;
    for (int e = 0; e < EE; ++e) {
      float pm = sum_probs[e] / (float)TT;
      float fm = (float)counts[e] / (float)(TT * KK);
      acc += pm * fm;
    }
    y[(size_t)TT * DOUT] = (float)EE * acc;
  }
}

// ---------------------------------------------------------------------------
extern "C" void kernel_launch(void* const* d_in, const int* in_sizes, int n_in,
                              void* d_out, int out_size, void* d_ws, size_t ws_size,
                              hipStream_t stream) {
  const float* x  = (const float*)d_in[0];
  const float* Wr = (const float*)d_in[1];
  const float* br = (const float*)d_in[2];
  const float* We = (const float*)d_in[3];
  const float* be = (const float*)d_in[4];
  float* y = (float*)d_out;

  // workspace: [counts:8 int][sum_probs:8 f32][pad -> 64 dwords]
  //            [etok: E*CAP int][egate: E*CAP f32]
  //            [xbf: T*D bf16][Webf: E*D*DOUT bf16]   (fast path only)
  int*   counts    = (int*)d_ws;
  float* sum_probs = (float*)d_ws + 8;
  int*   etok      = (int*)d_ws + 64;
  float* egate     = (float*)((int*)d_ws + 64 + EE * CAP);
  const size_t base_bytes = (size_t)(64 + 2 * EE * CAP) * 4;
  unsigned short* xbf  = (unsigned short*)((char*)d_ws + base_bytes);
  unsigned short* Webf = xbf + (size_t)TT * DD;
  const size_t need_fast =
      base_bytes + ((size_t)TT * DD + (size_t)EE * DD * DOUT) * 2;

  moe_zero<<<(TT * DOUT) / 2048, 256, 0, stream>>>(y, (int*)d_ws);
  moe_router<<<TT / 8, 256, 0, stream>>>(x, Wr, br, counts, sum_probs, etok, egate);

  if (ws_size >= need_fast) {
    moe_convert<<<((TT * DD) + (EE * DD * DOUT)) / 2048, 256, 0, stream>>>(
        x, We, xbf, Webf);
    moe_expert_gemm_bf16<<<dim3(EE * 64, DOUT / 128), 256, 0, stream>>>(
        xbf, Webf, be, counts, etok, egate, y);
  } else {
    moe_expert_gemm_f32<<<dim3(EE * 64, DOUT / 128), 256, 0, stream>>>(
        x, We, be, counts, etok, egate, y);
  }
  moe_aux<<<1, 32, 0, stream>>>(counts, sum_probs, y);
}
